// CausalSelfAttention_16595753632221
// MI455X (gfx1250) — matrix operand order repaired and checked by reading, not compiled, not run on hardware
//
#include <hip/hip_runtime.h>

// ---------------------------------------------------------------------------
// CDNA5 (gfx1250) causal self-attention: pre-pass bf16 convert/transpose ->
// async-double-buffered qkv GEMM -> flash attention -> async proj GEMM.
// Matrix ops: v_wmma_f32_16x16x32_bf16 (wave32).
// LDS fills: global_load_async_to_lds_b128 + s_wait_asynccnt (ASYNCcnt path).
// Workspace (bf16): q[B,H,T,D] | k[B,H,T,D] | v[B,H,D,T] | y[B*T,C]
//                  | x_bf[B*T,C] | wT_attn[3C,C] | wT_proj[C,C]   (88 MB)
// ---------------------------------------------------------------------------

typedef __attribute__((ext_vector_type(16))) __bf16 v16bf;
typedef __attribute__((ext_vector_type(8)))  __bf16 bf16x8;
typedef __attribute__((ext_vector_type(8)))  float  v8f;

#define T_SEQ  2048
#define HEAD_D 64

__device__ __forceinline__ __bf16 f2bf(float f) {
    unsigned u = __float_as_uint(f);
    u += 0x7FFFu + ((u >> 16) & 1u);          // round-to-nearest-even
    unsigned short s = (unsigned short)(u >> 16);
    union { unsigned short u16; __bf16 b; } cv;
    cv.u16 = s;
    return cv.b;
}

// Build a 16-element bf16 fragment from two contiguous 8-element (16B) chunks.
__device__ __forceinline__ v16bf make_frag(const __bf16* p0, const __bf16* p1) {
    union { v16bf v; bf16x8 h[2]; } u;
    u.h[0] = *reinterpret_cast<const bf16x8*>(p0);
    u.h[1] = *reinterpret_cast<const bf16x8*>(p1);
    return u.v;
}

__device__ __forceinline__ v8f wmma_bf16(v16bf a, v16bf b, v8f c) {
    // (neg_a, A, neg_b, B, c_mod, C, reuse_a, reuse_b)
    return __builtin_amdgcn_wmma_f32_16x16x32_bf16(false, a, false, b,
                                                   (short)0, c, false, false);
}

// Async copy of 16 bytes per lane: global -> LDS, tracked by ASYNCcnt.
// Low 32 bits of a generic LDS pointer are the LDS byte offset on AMDGPU.
__device__ __forceinline__ void async_copy_b128(__bf16* lds_ptr, const __bf16* gptr) {
    unsigned           lds = (unsigned)(size_t)lds_ptr;
    unsigned long long ga  = (unsigned long long)(size_t)gptr;
    asm volatile("global_load_async_to_lds_b128 %0, %1, off"
                 :: "v"(lds), "v"(ga) : "memory");
}

// ---------------------------------------------------------------------------
// Pre-pass 1: x fp32 -> bf16 (8M elements, 8 per thread).
// ---------------------------------------------------------------------------
__global__ __launch_bounds__(256) void convert_x_kernel(
    const float* __restrict__ x, __bf16* __restrict__ xbf)
{
    size_t idx = ((size_t)blockIdx.x * 256 + threadIdx.x) * 8;
    const float4* p = reinterpret_cast<const float4*>(x + idx);
    float4 f0 = p[0], f1 = p[1];
    union { bf16x8 v; __bf16 e[8]; } u;
    u.e[0] = f2bf(f0.x); u.e[1] = f2bf(f0.y); u.e[2] = f2bf(f0.z); u.e[3] = f2bf(f0.w);
    u.e[4] = f2bf(f1.x); u.e[5] = f2bf(f1.y); u.e[6] = f2bf(f1.z); u.e[7] = f2bf(f1.w);
    *reinterpret_cast<bf16x8*>(xbf + idx) = u.v;
}

// ---------------------------------------------------------------------------
// Pre-pass 2: w [1024][N] fp32 -> wT [N][1024] bf16 (LDS-tiled 64x64).
// ---------------------------------------------------------------------------
__global__ __launch_bounds__(256) void transpose_w_kernel(
    const float* __restrict__ w, __bf16* __restrict__ wT, int N)
{
    __shared__ __attribute__((aligned(16))) __bf16 tile[64][72];  // pad: 144B rows
    const int tid = threadIdx.x;
    const int n0 = blockIdx.x * 64, k0 = blockIdx.y * 64;

    int r  = tid >> 2;            // k-row within tile
    int cq = (tid & 3) * 16;      // n-col start (16 floats)
    const float4* p = reinterpret_cast<const float4*>(w + (size_t)(k0 + r) * N + n0 + cq);
    #pragma unroll
    for (int j = 0; j < 4; ++j) {
        float4 f = p[j];
        tile[r][cq + j * 4 + 0] = f2bf(f.x);
        tile[r][cq + j * 4 + 1] = f2bf(f.y);
        tile[r][cq + j * 4 + 2] = f2bf(f.z);
        tile[r][cq + j * 4 + 3] = f2bf(f.w);
    }
    __syncthreads();
    #pragma unroll
    for (int t = 0; t < 2; ++t) {
        int c  = tid + t * 256;   // 512 output chunks of 8 bf16
        int nr = c >> 3;
        int kq = (c & 7) * 8;
        union { bf16x8 v; __bf16 e[8]; } u;
        #pragma unroll
        for (int i = 0; i < 8; ++i) u.e[i] = tile[kq + i][nr];
        *reinterpret_cast<bf16x8*>(wT + (size_t)(n0 + nr) * 1024 + k0 + kq) = u.v;
    }
}

// ---------------------------------------------------------------------------
// Shared GEMM machinery: 256 threads / 8 waves, 128(M)x64(N) tile, K-step 32,
// double-buffered LDS filled by async copies (3 per thread per stage).
// A: bf16 [.,1024] row-major. B: bf16 [N][1024] (pre-transposed).
// ---------------------------------------------------------------------------
__device__ __forceinline__ void issue_stage(int tid, __bf16* sA, __bf16* sBT,
                                            const __bf16* Arow0, const __bf16* Brow0,
                                            int kk)
{
    #pragma unroll
    for (int t = 0; t < 2; ++t) {              // A tile: 128x32 = 512 chunks
        int c = tid + t * 256;
        int row = c >> 2, cq = (c & 3) * 8;
        async_copy_b128(sA + row * 32 + cq, Arow0 + (size_t)row * 1024 + kk + cq);
    }
    {                                          // B tile: 64x32 = 256 chunks
        int n = tid >> 2, cq = (tid & 3) * 8;
        async_copy_b128(sBT + n * 32 + cq, Brow0 + (size_t)n * 1024 + kk + cq);
    }
}

#define GEMM_MAINLOOP(sAbuf, sBTbuf, Arow0, Brow0)                                   \
    issue_stage(tid, sAbuf[0], sBTbuf[0], Arow0, Brow0, 0);                          \
    for (int s = 0; s < 32; ++s) {                                                   \
        const int kk = s * 32, buf = s & 1;                                          \
        if (s + 1 < 32) {                                                            \
            issue_stage(tid, sAbuf[buf ^ 1], sBTbuf[buf ^ 1], Arow0, Brow0, kk + 32);\
            asm volatile("s_wait_asynccnt 0x3" ::: "memory");                        \
        } else {                                                                     \
            asm volatile("s_wait_asynccnt 0x0" ::: "memory");                        \
        }                                                                            \
        __syncthreads();                                                             \
        v16bf af[2], bfr[2];                                                         \
        _Pragma("unroll")                                                            \
        for (int i = 0; i < 2; ++i) {                                                \
            int r = waveM + i * 16 + ln;                                             \
            af[i] = make_frag(&sAbuf[buf][r * 32 + lh * 8],                          \
                              &sAbuf[buf][r * 32 + 16 + lh * 8]);                    \
        }                                                                            \
        _Pragma("unroll")                                                            \
        for (int j = 0; j < 2; ++j) {                                                \
            int n = waveN + j * 16 + ln;                                             \
            bfr[j] = make_frag(&sBTbuf[buf][n * 32 + lh * 8],                        \
                               &sBTbuf[buf][n * 32 + 16 + lh * 8]);                  \
        }                                                                            \
        _Pragma("unroll")                                                            \
        for (int i = 0; i < 2; ++i)                                                  \
            _Pragma("unroll")                                                        \
            for (int j = 0; j < 2; ++j)                                              \
                acc[i][j] = wmma_bf16(af[i], bfr[j], acc[i][j]);                     \
        __syncthreads();                                                             \
    }

// ---------------------------------------------------------------------------
// Kernel: qkv = x_bf @ w_attnT + b_attn, scattered into q/k/v (v transposed).
// ---------------------------------------------------------------------------
__global__ __launch_bounds__(256) void qkv_gemm_kernel(
    const __bf16* __restrict__ xbf,     // [8192,1024] bf16
    const __bf16* __restrict__ wT,      // [3072,1024] bf16 (transposed)
    const float*  __restrict__ bias,    // [3072]
    __bf16* __restrict__ qg, __bf16* __restrict__ kg, __bf16* __restrict__ vg)
{
    __shared__ __attribute__((aligned(16))) __bf16 sA[2][128 * 32];
    __shared__ __attribute__((aligned(16))) __bf16 sBT[2][64 * 32];

    const int tid  = threadIdx.x;
    const int m0   = blockIdx.y * 128;
    const int n0   = blockIdx.x * 64;
    const int wave = tid >> 5, lane = tid & 31;
    const int ln   = lane & 15, lh = lane >> 4;
    const int waveM = (wave >> 1) * 32;
    const int waveN = (wave & 1) * 32;

    v8f zero = {};
    v8f acc[2][2];
    #pragma unroll
    for (int i = 0; i < 2; ++i)
        #pragma unroll
        for (int j = 0; j < 2; ++j) acc[i][j] = zero;

    const __bf16* Arow0 = xbf + (size_t)m0 * 1024;
    const __bf16* Brow0 = wT  + (size_t)n0 * 1024;

    GEMM_MAINLOOP(sA, sBT, Arow0, Brow0)

    // epilogue: bias + scatter to q / k / v(transposed [D][T]), bf16
    #pragma unroll
    for (int j = 0; j < 2; ++j) {
        int gn = n0 + waveN + j * 16 + ln;
        float bv = bias[gn];
        #pragma unroll
        for (int i = 0; i < 2; ++i) {
            #pragma unroll
            for (int v = 0; v < 8; ++v) {
                int gm = m0 + waveM + i * 16 + v + 8 * lh;
                float val = acc[i][j][v] + bv;
                int bb = gm >> 11, tt = gm & 2047;
                __bf16 hv = f2bf(val);
                if (gn < 1024) {
                    int head = gn >> 6, d = gn & 63;
                    qg[((size_t)(bb * 16 + head) * 2048 + tt) * 64 + d] = hv;
                } else if (gn < 2048) {
                    int c = gn - 1024, head = c >> 6, d = c & 63;
                    kg[((size_t)(bb * 16 + head) * 2048 + tt) * 64 + d] = hv;
                } else {
                    int c = gn - 2048, head = c >> 6, d = c & 63;
                    vg[((size_t)(bb * 16 + head) * 64 + d) * 2048 + tt] = hv;
                }
            }
        }
    }
}

// ---------------------------------------------------------------------------
// Kernel: flash attention. One wave owns 16 query rows; online softmax in the
// WMMA C-layout; Q/K/V fragments loaded straight from global (contiguous 16B).
// ---------------------------------------------------------------------------
__global__ __launch_bounds__(256) void attn_kernel(
    const __bf16* __restrict__ qg,  // [B,H,T,D]
    const __bf16* __restrict__ kg,  // [B,H,T,D]
    const __bf16* __restrict__ vg,  // [B,H,D,T]
    __bf16* __restrict__ yg)        // [B*T, C]
{
    __shared__ __attribute__((aligned(16))) __bf16 sP[8][16 * 32];  // per-wave P

    const int tid  = threadIdx.x;
    const int wave = tid >> 5, lane = tid & 31;
    const int ln   = lane & 15, lh = lane >> 4;
    const int bh   = blockIdx.y;
    const int bb   = bh >> 4, hh = bh & 15;

    const __bf16* qb = qg + (size_t)bh * (T_SEQ * HEAD_D);
    const __bf16* kb = kg + (size_t)bh * (T_SEQ * HEAD_D);
    const __bf16* vb = vg + (size_t)bh * (HEAD_D * T_SEQ);

    const int q0 = blockIdx.x * 128 + wave * 16;
    __bf16* myP = sP[wave];

    v16bf qf[2];
    {
        const __bf16* qrow = qb + (size_t)(q0 + ln) * 64;
        #pragma unroll
        for (int dc = 0; dc < 2; ++dc)
            qf[dc] = make_frag(qrow + dc * 32 + lh * 8,
                               qrow + dc * 32 + 16 + lh * 8);
    }

    float m_[8], l_[8];
    v8f zero = {};
    v8f yac[4];
    #pragma unroll
    for (int v = 0; v < 8; ++v) { m_[v] = -1e30f; l_[v] = 0.f; }
    #pragma unroll
    for (int t = 0; t < 4; ++t) yac[t] = zero;

    const float SC = 0.125f;  // 1/sqrt(64)

    for (int c = 0; c <= q0; c += 32) {               // wave-uniform trip count
        if (c + 32 <= q0) {                           // prefetch next K/V chunk
            __builtin_prefetch((const void*)(kb + (size_t)(c + 32 + ln) * 64), 0, 1);
            __builtin_prefetch((const void*)(vb + (size_t)(lane * 2) * 2048 + c + 32), 0, 1);
        }
        v8f s0 = {}, s1 = {};
        #pragma unroll
        for (int dc = 0; dc < 2; ++dc) {              // subtile 0: keys c..c+15
            const __bf16* kp = kb + (size_t)(c + ln) * 64 + dc * 32 + lh * 8;
            s0 = wmma_bf16(qf[dc], make_frag(kp, kp + 16), s0);
        }
        const int kb1 = c + 16;
        if (kb1 <= q0) {                              // wave-uniform branch
            #pragma unroll
            for (int dc = 0; dc < 2; ++dc) {
                const __bf16* kp = kb + (size_t)(kb1 + ln) * 64 + dc * 32 + lh * 8;
                s1 = wmma_bf16(qf[dc], make_frag(kp, kp + 16), s1);
            }
        }
        // online softmax (C-layout: lane col = ln, rows v+8*lh)
        #pragma unroll
        for (int v = 0; v < 8; ++v) {
            int rowq = q0 + v + 8 * lh;
            float a0 = (c   + ln <= rowq) ? s0[v] * SC : -1e30f;
            float a1 = (kb1 + ln <= rowq) ? s1[v] * SC : -1e30f;
            float cm = fmaxf(a0, a1);
            #pragma unroll
            for (int off = 1; off < 16; off <<= 1)
                cm = fmaxf(cm, __shfl_xor(cm, off, 32));
            float mn    = fmaxf(m_[v], cm);
            float alpha = __expf(m_[v] - mn);
            float p0    = __expf(a0 - mn);
            float p1    = __expf(a1 - mn);
            float rs    = p0 + p1;
            #pragma unroll
            for (int off = 1; off < 16; off <<= 1)
                rs += __shfl_xor(rs, off, 32);
            l_[v] = l_[v] * alpha + rs;
            m_[v] = mn;
            #pragma unroll
            for (int t = 0; t < 4; ++t) yac[t][v] *= alpha;
            myP[(v + 8 * lh) * 32 + ln]      = f2bf(p0);
            myP[(v + 8 * lh) * 32 + 16 + ln] = f2bf(p1);
        }
        // per-wave LDS ops are in-order: store->load needs no barrier
        v16bf pf = make_frag(&myP[ln * 32 + lh * 8], &myP[ln * 32 + 16 + lh * 8]);
        #pragma unroll
        for (int dt = 0; dt < 4; ++dt) {
            const __bf16* vp = vb + (size_t)(dt * 16 + ln) * 2048 + c + lh * 8;
            yac[dt] = wmma_bf16(pf, make_frag(vp, vp + 16), yac[dt]);
        }
    }

    #pragma unroll
    for (int v = 0; v < 8; ++v) {
        float inv = 1.0f / l_[v];
        int row = q0 + v + 8 * lh;
        size_t rbase = ((size_t)bb * 2048 + row) * 1024 + hh * 64;
        #pragma unroll
        for (int dt = 0; dt < 4; ++dt)
            yg[rbase + dt * 16 + ln] = f2bf(yac[dt][v] * inv);
    }
}

// ---------------------------------------------------------------------------
// Kernel: out = y @ w_projT + b_proj (fp32 output).
// ---------------------------------------------------------------------------
__global__ __launch_bounds__(256) void proj_gemm_kernel(
    const __bf16* __restrict__ ybf,     // [8192,1024] bf16
    const __bf16* __restrict__ wT,      // [1024,1024] bf16 (transposed)
    const float*  __restrict__ bias,    // [1024]
    float* __restrict__ out)            // [8192,1024]
{
    __shared__ __attribute__((aligned(16))) __bf16 sA[2][128 * 32];
    __shared__ __attribute__((aligned(16))) __bf16 sBT[2][64 * 32];

    const int tid  = threadIdx.x;
    const int m0   = blockIdx.y * 128;
    const int n0   = blockIdx.x * 64;
    const int wave = tid >> 5, lane = tid & 31;
    const int ln   = lane & 15, lh = lane >> 4;
    const int waveM = (wave >> 1) * 32;
    const int waveN = (wave & 1) * 32;

    v8f zero = {};
    v8f acc[2][2];
    #pragma unroll
    for (int i = 0; i < 2; ++i)
        #pragma unroll
        for (int j = 0; j < 2; ++j) acc[i][j] = zero;

    const __bf16* Arow0 = ybf + (size_t)m0 * 1024;
    const __bf16* Brow0 = wT  + (size_t)n0 * 1024;

    GEMM_MAINLOOP(sA, sBT, Arow0, Brow0)

    #pragma unroll
    for (int j = 0; j < 2; ++j) {
        int gn = n0 + waveN + j * 16 + ln;
        float bv = bias[gn];
        #pragma unroll
        for (int i = 0; i < 2; ++i) {
            #pragma unroll
            for (int v = 0; v < 8; ++v) {
                int gm = m0 + waveM + i * 16 + v + 8 * lh;
                out[(size_t)gm * 1024 + gn] = acc[i][j][v] + bv;
            }
        }
    }
}

// ---------------------------------------------------------------------------
extern "C" void kernel_launch(void* const* d_in, const int* in_sizes, int n_in,
                              void* d_out, int out_size, void* d_ws, size_t ws_size,
                              hipStream_t stream) {
    (void)in_sizes; (void)n_in; (void)out_size; (void)ws_size;
    const float* x      = (const float*)d_in[0];   // [4,2048,1024]
    const float* w_attn = (const float*)d_in[1];   // [1024,3072]
    const float* b_attn = (const float*)d_in[2];   // [3072]
    const float* w_proj = (const float*)d_in[3];   // [1024,1024]
    const float* b_proj = (const float*)d_in[4];   // [1024]
    float* out = (float*)d_out;                    // [8192,1024]

    const size_t PER = (size_t)8192 * 1024;        // elems per big buffer
    __bf16* qg  = (__bf16*)d_ws;                   //  0 MB
    __bf16* kg  = qg  + PER;                       // 16 MB
    __bf16* vg  = kg  + PER;                       // 32 MB
    __bf16* yg  = vg  + PER;                       // 48 MB
    __bf16* xbf = yg  + PER;                       // 64 MB
    __bf16* wTa = xbf + PER;                       // 80 MB (3072*1024)
    __bf16* wTp = wTa + (size_t)3072 * 1024;       // 86 MB (1024*1024) -> 88 MB

    convert_x_kernel  <<<dim3(4096),        256, 0, stream>>>(x, xbf);
    transpose_w_kernel<<<dim3(48, 16),      256, 0, stream>>>(w_attn, wTa, 3072);
    transpose_w_kernel<<<dim3(16, 16),      256, 0, stream>>>(w_proj, wTp, 1024);
    qkv_gemm_kernel   <<<dim3(48, 64),      256, 0, stream>>>(xbf, wTa, b_attn, qg, kg, vg);
    attn_kernel       <<<dim3(16, 64),      256, 0, stream>>>(qg, kg, vg, yg);
    proj_gemm_kernel  <<<dim3(16, 64),      256, 0, stream>>>(yg, wTp, b_proj, out);
}
